// Encoder_TRANSFORMER_7138235646285
// MI455X (gfx1250) — matrix operand order repaired and checked by reading, not compiled, not run on hardware
//
#include <hip/hip_runtime.h>
#include <math.h>

// ---------------- problem constants ----------------
constexpr int Bc = 2, Sc = 25, Nc = 5000, Ec = 15000;
constexpr int Gc = Bc * Sc;               // 50 graphs
constexpr int NNc = Gc * Nc;              // 250000 nodes
constexpr int GEc = Gc * Ec;              // 750000 edges
constexpr int E1c = 64, E2c = 128, E3c = 256, CDIMc = 12, Zc = 32;
constexpr int Tc = Sc + 2;                // 27 tokens
constexpr int NTOKc = Tc * Bc;            // 54 (t,b) tokens

typedef _Float16 v16h __attribute__((ext_vector_type(16)));
typedef float    v8f  __attribute__((ext_vector_type(8)));
typedef float    v4f  __attribute__((ext_vector_type(4)));

static inline int cdiv(long long a, long long b) { return (int)((a + b - 1) / b); }

// ---------------- 1. skelEmbedding: relu(v @ W_emb^T + b), write f16 ----------------
__global__ void embed_kernel(const float* __restrict__ v, const float* __restrict__ W,
                             const float* __restrict__ b, _Float16* __restrict__ X) {
  int idx = blockIdx.x * blockDim.x + threadIdx.x;
  if (idx >= NNc * E1c) return;
  int node = idx >> 6, o = idx & 63;
  const float* vp = v + (size_t)node * 3;
  float acc = b[o] + vp[0] * W[o * 3 + 0] + vp[1] * W[o * 3 + 1] + vp[2] * W[o * 3 + 2];
  X[idx] = (_Float16)fmaxf(acc, 0.0f);
}

// ---------------- 2. degrees (shared by both GCN layers) ----------------
__global__ void deg_init(float* deg) {
  int i = blockIdx.x * blockDim.x + threadIdx.x;
  if (i < NNc) deg[i] = 1.0f;               // +1 self loop
}
__global__ void deg_count(const int* __restrict__ edges, float* deg) {
  int e = blockIdx.x * blockDim.x + threadIdx.x;
  if (e >= GEc) return;
  int g = e / Ec;
  int dst = edges[(size_t)e * 2 + 1];
  atomicAdd(&deg[g * Nc + dst], 1.0f);
}
__global__ void deg_rsqrt(float* deg) {
  int i = blockIdx.x * blockDim.x + threadIdx.x;
  if (i < NNc) deg[i] = rsqrtf(deg[i]);
}

// ---------------- 3. pack weights [K x NOUT] f32 -> per-lane WMMA B layout, f16 ----------------
// B 32x16 f16 layout: lane (n = lane&15, khalf = lane>>4) holds 16 contiguous K values:
//   b[j] = B[kstep*32 + khalf*16 + j][ntile*16 + n]
__global__ void pack_b_kernel(const float* __restrict__ W, _Float16* __restrict__ Bp,
                              int K, int NOUT) {
  int p = blockIdx.x * blockDim.x + threadIdx.x;
  if (p >= K * NOUT) return;
  int NT = NOUT / 16;
  int j = p & 15;
  int lane = (p >> 4) & 31;
  int nt = (p >> 9) % NT;
  int ks = p / (512 * NT);
  int n = lane & 15, khalf = lane >> 4;
  int k = ks * 32 + khalf * 16 + j;
  Bp[p] = (_Float16)W[(size_t)k * NOUT + nt * 16 + n];
}

// ---------------- 4. WMMA GEMM: H[M x NOUT] = A[M x K](f16) @ W(packed f16), f32 acc ----------
// One wave per 16-row tile, 8 waves / block. A-matrix 16x32 f16 per-lane layout:
//   a[0..7]  = A[m][ks*32 + khalf*16 + 0..7]
//   a[8..15] = A[m][ks*32 + khalf*16 + 8..15]
template <int K, int NOUT>
__global__ __launch_bounds__(256) void gcn_gemm_wmma(const _Float16* __restrict__ A,
                                                     const _Float16* __restrict__ Bp,
                                                     float* __restrict__ H, int M) {
  constexpr int NT = NOUT / 16;
  constexpr int KS = K / 32;
  int wave = threadIdx.x >> 5;
  int lane = threadIdx.x & 31;
  int tile = blockIdx.x * 8 + wave;
  if (tile * 16 >= M) return;
  int mrow = tile * 16 + (lane & 15);
  int khalf = lane >> 4;

  v8f acc[NT];
#pragma unroll
  for (int i = 0; i < NT; i++) acc[i] = v8f{0.f, 0.f, 0.f, 0.f, 0.f, 0.f, 0.f, 0.f};

#pragma unroll
  for (int ks = 0; ks < KS; ks++) {
    const _Float16* ap = A + (size_t)mrow * K + ks * 32 + khalf * 16;
    v16h a;
    ((v4f*)&a)[0] = *(const v4f*)(ap);        // 16B: K+0..7
    ((v4f*)&a)[1] = *(const v4f*)(ap + 8);    // 16B: K+8..15
#pragma unroll
    for (int nt = 0; nt < NT; nt++) {
      const _Float16* bp = Bp + ((size_t)(ks * NT + nt) * 32 + lane) * 16;
      v16h bmat;
      ((v4f*)&bmat)[0] = *(const v4f*)(bp);
      ((v4f*)&bmat)[1] = *(const v4f*)(bp + 8);
      acc[nt] = __builtin_amdgcn_wmma_f32_16x16x32_f16(
          /*neg_a=*/false, a, /*neg_b=*/false, bmat,
          /*c_mod=*/(short)0, acc[nt], /*reuse_a=*/false, /*reuse_b=*/false);
    }
  }
  // C/D layout: lane holds column n = lane&15; VGPR r -> row r + 8*khalf
  int ncol = lane & 15;
  int mbase = tile * 16 + 8 * khalf;
#pragma unroll
  for (int nt = 0; nt < NT; nt++) {
#pragma unroll
    for (int r = 0; r < 8; r++) {
      H[(size_t)(mbase + r) * NOUT + nt * 16 + ncol] = acc[nt][r];
    }
  }
}

// ---------------- 5. AGG init: self-loop term + bias ----------------
template <int NOUT>
__global__ void agg_init_kernel(const float* __restrict__ H, const float* __restrict__ dinv,
                                const float* __restrict__ bias, float* __restrict__ AGG) {
  size_t idx = (size_t)blockIdx.x * blockDim.x + threadIdx.x;
  if (idx >= (size_t)NNc * NOUT) return;
  int i = (int)(idx / NOUT);
  int f = (int)(idx % NOUT);
  float di = dinv[i];
  AGG[idx] = H[idx] * di * di + bias[f];
}

// ---------------- 6. edge scatter: AGG[dst] += H[src] * dinv[src]*dinv[dst] ----------------
template <int NOUT>
__global__ void scatter_kernel(const int* __restrict__ edges, const float* __restrict__ dinv,
                               const float* __restrict__ H, float* __restrict__ AGG) {
  constexpr int C = NOUT / 4;
  size_t idx = (size_t)blockIdx.x * blockDim.x + threadIdx.x;
  if (idx >= (size_t)GEc * C) return;
  int e = (int)(idx / C);
  int c = (int)(idx % C) * 4;
  int g = e / Ec;
  int src = edges[(size_t)e * 2 + 0];
  int dst = edges[(size_t)e * 2 + 1];
  float coef = dinv[g * Nc + src] * dinv[g * Nc + dst];
  const float* hp = H + ((size_t)g * Nc + src) * NOUT + c;
  v4f hv = *(const v4f*)hp;
  float* ap = AGG + ((size_t)g * Nc + dst) * NOUT + c;
#pragma unroll
  for (int k = 0; k < 4; k++) atomicAdd(ap + k, hv[k] * coef);
}

// ---------------- 7. leaky_relu -> f16 (feeds next GEMM) ----------------
__global__ void leaky_to_f16(const float* __restrict__ AGG, _Float16* __restrict__ X, size_t n) {
  size_t idx = (size_t)blockIdx.x * blockDim.x + threadIdx.x;
  if (idx >= n) return;
  float x = AGG[idx];
  x = (x > 0.f) ? x : 0.15f * x;
  X[idx] = (_Float16)x;
}

// ---------------- 8. leaky_relu + max-pool over nodes ----------------
__global__ __launch_bounds__(256) void pool_kernel(const float* __restrict__ AGG,
                                                   float* __restrict__ pooled) {
  int g = blockIdx.x;     // 50
  int f = threadIdx.x;    // 256
  float m = -INFINITY;
  for (int n = 0; n < Nc; n++) {
    float x = AGG[((size_t)g * Nc + n) * E3c + f];
    x = (x > 0.f) ? x : 0.15f * x;
    m = fmaxf(m, x);
  }
  pooled[g * E3c + f] = m;
}

// ---------------- 9. FC: relu(concat(pooled, con) @ W_fc + b_fc) ----------------
__global__ void fc_kernel(const float* __restrict__ pooled, const float* __restrict__ con,
                          const float* __restrict__ Wfc, const float* __restrict__ bfc,
                          float* __restrict__ xfc) {
  int g = blockIdx.x;     // 50
  int z = threadIdx.x;    // 32
  int b = g / Sc;
  float acc = bfc[z];
  for (int j = 0; j < E3c; j++) acc += pooled[g * E3c + j] * Wfc[j * Zc + z];
  for (int j = 0; j < CDIMc; j++) acc += con[b * CDIMc + j] * Wfc[(E3c + j) * Zc + z];
  xfc[g * Zc + z] = fmaxf(acc, 0.0f);
}

// ---------------- 10. build xseq [27,2,32] + positional encoding ----------------
__global__ void build_xseq(const float* __restrict__ xfc, const float* __restrict__ muQ,
                           const float* __restrict__ sigQ, float* __restrict__ xseq) {
  int i = blockIdx.x * blockDim.x + threadIdx.x;
  if (i >= Tc * Bc * Zc) return;
  int z = i % Zc;
  int b = (i / Zc) % Bc;
  int t = i / (Zc * Bc);
  float val;
  if (t == 0) val = muQ[z];
  else if (t == 1) val = sigQ[z];
  else val = xfc[(b * Sc + (t - 2)) * Zc + z];
  float w = expf(-logf(10000.0f) * (float)(2 * (z / 2)) / (float)Zc);
  float ang = (float)t * w;
  val += (z & 1) ? cosf(ang) : sinf(ang);
  xseq[i] = val;
}

// ---------------- 11. 4-layer transformer encoder, single block, LDS resident ----------------
__global__ __launch_bounds__(256) void transformer_kernel(
    const float* __restrict__ xin, const float* __restrict__ Wqkv, const float* __restrict__ bqkv,
    const float* __restrict__ Wo, const float* __restrict__ bo,
    const float* __restrict__ ln1s, const float* __restrict__ ln1b,
    const float* __restrict__ ln2s, const float* __restrict__ ln2b,
    const float* __restrict__ W1, const float* __restrict__ b1,
    const float* __restrict__ W2, const float* __restrict__ b2,
    float* __restrict__ out) {
  __shared__ float sx[NTOKc * Zc];        // 1728  current activations, [tb][z], tb=t*2+b
  __shared__ float sqkv[NTOKc * 3 * Zc];  // 5184
  __shared__ float sA[5832];              // att scores / post-attn tmp / FF hidden chunk
  __shared__ float sO[NTOKc * Zc];        // 1728  attn-out / FF-out (+residual)
  int tid = threadIdx.x;

  for (int i = tid; i < NTOKc * Zc; i += 256) sx[i] = xin[i];
  __syncthreads();

  for (int l = 0; l < 4; l++) {
    const float* wq = Wqkv + (size_t)l * 96 * 32;
    const float* bq = bqkv + l * 96;
    // qkv = x @ Wqkv^T + b
    for (int i = tid; i < NTOKc * 96; i += 256) {
      int tb = i / 96, o = i % 96;
      float acc = bq[o];
      for (int z = 0; z < 32; z++) acc += sx[tb * 32 + z] * wq[o * 32 + z];
      sqkv[i] = acc;
    }
    __syncthreads();
    // scores [b,h,t,s]
    for (int i = tid; i < 2 * 4 * 27 * 27; i += 256) {
      int s = i % 27, t = (i / 27) % 27, h = (i / 729) % 4, b = i / 2916;
      float acc = 0.f;
      for (int d = 0; d < 8; d++)
        acc += sqkv[(t * 2 + b) * 96 + h * 8 + d] * sqkv[(s * 2 + b) * 96 + 32 + h * 8 + d];
      sA[i] = acc * 0.35355339059327373f;   // 1/sqrt(8)
    }
    __syncthreads();
    // softmax over s (216 rows)
    if (tid < 216) {
      float* row = sA + tid * 27;
      float m = row[0];
      for (int s = 1; s < 27; s++) m = fmaxf(m, row[s]);
      float sum = 0.f;
      for (int s = 0; s < 27; s++) { float e = expf(row[s] - m); row[s] = e; sum += e; }
      float inv = 1.0f / sum;
      for (int s = 0; s < 27; s++) row[s] *= inv;
    }
    __syncthreads();
    // o = att @ v
    for (int i = tid; i < NTOKc * Zc; i += 256) {
      int tb = i / 32, zz = i % 32, t = tb / 2, b = tb % 2, h = zz / 8, d = zz % 8;
      const float* arow = sA + ((b * 4 + h) * 27 + t) * 27;
      float acc = 0.f;
      for (int s = 0; s < 27; s++) acc += arow[s] * sqkv[(s * 2 + b) * 96 + 64 + h * 8 + d];
      sO[i] = acc;
    }
    __syncthreads();
    // out-proj + residual -> sA (tmp)
    const float* wo = Wo + (size_t)l * 32 * 32;
    const float* bov = bo + l * 32;
    for (int i = tid; i < NTOKc * Zc; i += 256) {
      int tb = i / 32, z = i % 32;
      float acc = bov[z];
      for (int j = 0; j < 32; j++) acc += sO[tb * 32 + j] * wo[z * 32 + j];
      sA[i] = sx[i] + acc;
    }
    __syncthreads();
    // LN1 -> sx
    if (tid < NTOKc) {
      float m = 0.f;
      for (int z = 0; z < 32; z++) m += sA[tid * 32 + z];
      m *= (1.0f / 32);
      float var = 0.f;
      for (int z = 0; z < 32; z++) { float d = sA[tid * 32 + z] - m; var += d * d; }
      var *= (1.0f / 32);
      float inv = rsqrtf(var + 1e-5f);
      for (int z = 0; z < 32; z++)
        sx[tid * 32 + z] = (sA[tid * 32 + z] - m) * inv * ln1s[l * 32 + z] + ln1b[l * 32 + z];
    }
    __syncthreads();
    // FF, 4-token chunks: hidden in sA (4*1024), out+residual in sO
    const float* w1 = W1 + (size_t)l * 1024 * 32;
    const float* b1v = b1 + l * 1024;
    const float* w2 = W2 + (size_t)l * 32 * 1024;
    const float* b2v = b2 + l * 32;
    for (int c0 = 0; c0 < NTOKc; c0 += 4) {
      int ntok = (NTOKc - c0 < 4) ? (NTOKc - c0) : 4;
      for (int i = tid; i < ntok * 1024; i += 256) {
        int tt = c0 + i / 1024, j = i % 1024;
        float acc = b1v[j];
        for (int z = 0; z < 32; z++) acc += sx[tt * 32 + z] * w1[j * 32 + z];
        sA[i] = 0.5f * acc * (1.0f + erff(acc * 0.7071067811865475f));  // exact gelu
      }
      __syncthreads();
      for (int i = tid; i < ntok * 32; i += 256) {
        int tt = c0 + i / 32, z = i % 32;
        const float* hrow = sA + (i / 32) * 1024;
        float acc = b2v[z];
        for (int j = 0; j < 1024; j++) acc += hrow[j] * w2[z * 1024 + j];
        sO[tt * 32 + z] = sx[tt * 32 + z] + acc;
      }
      __syncthreads();
    }
    // LN2 -> sx
    if (tid < NTOKc) {
      float m = 0.f;
      for (int z = 0; z < 32; z++) m += sO[tid * 32 + z];
      m *= (1.0f / 32);
      float var = 0.f;
      for (int z = 0; z < 32; z++) { float d = sO[tid * 32 + z] - m; var += d * d; }
      var *= (1.0f / 32);
      float inv = rsqrtf(var + 1e-5f);
      for (int z = 0; z < 32; z++)
        sx[tid * 32 + z] = (sO[tid * 32 + z] - m) * inv * ln2s[l * 32 + z] + ln2b[l * 32 + z];
    }
    __syncthreads();
  }
  // outputs: xseq[0] (64), xseq[1] (64), full xseq (1728)
  for (int i = tid; i < 64; i += 256) out[i] = sx[i];            // t=0 -> tb 0..1
  for (int i = tid; i < 64; i += 256) out[64 + i] = sx[64 + i];  // t=1 -> tb 2..3
  for (int i = tid; i < NTOKc * Zc; i += 256) out[128 + i] = sx[i];
}

// ---------------- host launcher ----------------
extern "C" void kernel_launch(void* const* d_in, const int* in_sizes, int n_in,
                              void* d_out, int out_size, void* d_ws, size_t ws_size,
                              hipStream_t stream) {
  const float* v      = (const float*)d_in[0];
  const int*   edges  = (const int*)d_in[1];
  const float* con    = (const float*)d_in[2];
  const float* W_emb  = (const float*)d_in[3];
  const float* b_emb  = (const float*)d_in[4];
  const float* W_g1   = (const float*)d_in[5];
  const float* b_g1   = (const float*)d_in[6];
  const float* W_g2   = (const float*)d_in[7];
  const float* b_g2   = (const float*)d_in[8];
  const float* W_fc   = (const float*)d_in[9];
  const float* b_fc   = (const float*)d_in[10];
  const float* muQ    = (const float*)d_in[11];
  const float* sigQ   = (const float*)d_in[12];
  const float* Wqkv   = (const float*)d_in[13];
  const float* bqkv   = (const float*)d_in[14];
  const float* Wo     = (const float*)d_in[15];
  const float* bo     = (const float*)d_in[16];
  const float* ln1s   = (const float*)d_in[17];
  const float* ln1b   = (const float*)d_in[18];
  const float* ln2s   = (const float*)d_in[19];
  const float* ln2b   = (const float*)d_in[20];
  const float* W1     = (const float*)d_in[21];
  const float* b1v    = (const float*)d_in[22];
  const float* W2     = (const float*)d_in[23];
  const float* b2v    = (const float*)d_in[24];

  char* ws = (char*)d_ws;
  size_t off = 0;
  auto take = [&](size_t bytes) -> void* {
    void* p = ws + off;
    off += (bytes + 255) & ~(size_t)255;
    return p;
  };
  _Float16* Xh     = (_Float16*)take((size_t)NNc * 128 * sizeof(_Float16)); // 64 MB
  float*    Hbuf   = (float*)take((size_t)NNc * 256 * sizeof(float));       // 256 MB
  float*    AGG    = (float*)take((size_t)NNc * 256 * sizeof(float));       // 256 MB
  float*    dinv   = (float*)take((size_t)NNc * sizeof(float));
  _Float16* Bp     = (_Float16*)take((size_t)128 * 256 * sizeof(_Float16));
  float*    pooled = (float*)take((size_t)Gc * 256 * sizeof(float));
  float*    xfc    = (float*)take((size_t)Gc * 32 * sizeof(float));
  float*    xseqb  = (float*)take((size_t)Tc * Bc * Zc * sizeof(float));

  const int TPB = 256;
  const int gemmBlocks = cdiv(NNc / 16, 8);   // 16-row tiles, 8 waves/block

  // embedding -> f16
  embed_kernel<<<cdiv((long long)NNc * E1c, TPB), TPB, 0, stream>>>(v, W_emb, b_emb, Xh);

  // degrees (shared across both GCN layers)
  deg_init<<<cdiv(NNc, TPB), TPB, 0, stream>>>(dinv);
  deg_count<<<cdiv(GEc, TPB), TPB, 0, stream>>>(edges, dinv);
  deg_rsqrt<<<cdiv(NNc, TPB), TPB, 0, stream>>>(dinv);

  // ---- GCN layer 1: 64 -> 128 ----
  pack_b_kernel<<<cdiv(E1c * E2c, TPB), TPB, 0, stream>>>(W_g1, Bp, E1c, E2c);
  gcn_gemm_wmma<64, 128><<<gemmBlocks, TPB, 0, stream>>>(Xh, Bp, Hbuf, NNc);
  agg_init_kernel<128><<<cdiv((long long)NNc * 128, TPB), TPB, 0, stream>>>(Hbuf, dinv, b_g1, AGG);
  scatter_kernel<128><<<cdiv((long long)GEc * 32, TPB), TPB, 0, stream>>>(edges, dinv, Hbuf, AGG);
  leaky_to_f16<<<cdiv((long long)NNc * 128, TPB), TPB, 0, stream>>>(AGG, Xh, (size_t)NNc * 128);

  // ---- GCN layer 2: 128 -> 256 ----
  pack_b_kernel<<<cdiv(E2c * E3c, TPB), TPB, 0, stream>>>(W_g2, Bp, E2c, E3c);
  gcn_gemm_wmma<128, 256><<<gemmBlocks, TPB, 0, stream>>>(Xh, Bp, Hbuf, NNc);
  agg_init_kernel<256><<<cdiv((long long)NNc * 256, TPB), TPB, 0, stream>>>(Hbuf, dinv, b_g2, AGG);
  scatter_kernel<256><<<cdiv((long long)GEc * 64, TPB), TPB, 0, stream>>>(edges, dinv, Hbuf, AGG);

  // pool (fused leaky), FC, sequence build, transformer
  pool_kernel<<<Gc, TPB, 0, stream>>>(AGG, pooled);
  fc_kernel<<<Gc, Zc, 0, stream>>>(pooled, con, W_fc, b_fc, xfc);
  build_xseq<<<cdiv(Tc * Bc * Zc, TPB), TPB, 0, stream>>>(xfc, muQ, sigQ, xseqb);
  transformer_kernel<<<1, TPB, 0, stream>>>(xseqb, Wqkv, bqkv, Wo, bo, ln1s, ln1b,
                                            ln2s, ln2b, W1, b1v, W2, b2v, (float*)d_out);
}
